// SparseResBlockSE_17849884082698
// MI455X (gfx1250) — matrix operand order repaired and checked by reading, not compiled, not run on hardware
//
#include <hip/hip_runtime.h>
#include <cstddef>
#include <cstdint>

// ---------------------------------------------------------------------------
// MI455X (gfx1250) implementation: bf16 WMMA (v_wmma_f32_16x16x32_bf16) for
// the two gather-GEMM convolutions and the skip GEMM; fp32 everywhere else.
// Conv kernel: 32-row M tile, dual accumulators per wave -> each B fragment
// (L2-resident swizzled weights) feeds 2 WMMAs, halving L2 B-traffic.
// ---------------------------------------------------------------------------

#define NPTS   262144      // B*M_PER_B
#define HH     512
#define WW     512
#define NB     8
#define HWSZ   (HH*WW)
#define COUT   128

typedef __attribute__((ext_vector_type(16))) __bf16 v16bf;
typedef __attribute__((ext_vector_type(8)))  float  v8f;

struct B32x2 { uint4 lo, hi; };   // 32 bytes == one v16bf fragment

__device__ __forceinline__ unsigned f2bf1(float f) {
    unsigned u = __builtin_bit_cast(unsigned, f);
    return (u + 0x7fffu + ((u >> 16) & 1u)) >> 16;       // RNE f32 -> bf16
}
__device__ __forceinline__ unsigned pack2(float lo, float hi) {
    return f2bf1(lo) | (f2bf1(hi) << 16);
}

// ---------------------------------------------------------------- rulebook --
__global__ __launch_bounds__(256) void k_init_grid(int* grid) {
    int i = blockIdx.x * 256 + threadIdx.x;
    if (i < NB * HWSZ) grid[i] = -1;
}
__global__ __launch_bounds__(256) void k_init_stats(float* s, int n) {
    for (int i = threadIdx.x; i < n; i += 256) s[i] = 0.0f;
}
__global__ __launch_bounds__(256) void k_scatter(const int* __restrict__ idx, int* __restrict__ grid) {
    int n = blockIdx.x * 256 + threadIdx.x;
    if (n >= NPTS) return;
    int b = idx[3*n], y = idx[3*n+1], x = idx[3*n+2];
    grid[b * HWSZ + y * WW + x] = n;
}
__global__ __launch_bounds__(256) void k_rulebook(const int* __restrict__ idx,
                                                  const int* __restrict__ grid,
                                                  int* __restrict__ nbr) {
    int n = blockIdx.x * 256 + threadIdx.x;
    if (n >= NPTS) return;
    int b = idx[3*n], y = idx[3*n+1], x = idx[3*n+2];
    int k = 0;
    #pragma unroll
    for (int dy = -1; dy <= 1; ++dy)
        #pragma unroll
        for (int dx = -1; dx <= 1; ++dx, ++k) {
            int ny = y + dy, nx = x + dx;
            int nid = -1;
            if (ny >= 0 && ny < HH && nx >= 0 && nx < WW)
                nid = grid[b * HWSZ + ny * WW + nx];
            nbr[(size_t)n * 9 + k] = nid;
        }
}

// --------------------------------------------------- weight swizzle to bf16 --
// B-fragment layout (16-bit B, 32x16): lane l = h*16 + n holds column n,
// elements e=0..15 map to K = kc*32 + 16*h + e. Fragment id = (k*KC+kc)*8+nt.
// Stored fragment-contiguous: dst[frag*512 + l*16 + e].
__global__ __launch_bounds__(256) void k_swizzle(const float* __restrict__ w,
                                                 unsigned short* __restrict__ dst,
                                                 int KC, int cin, int total) {
    int tid = blockIdx.x * 256 + threadIdx.x;
    if (tid >= total) return;
    int e    = tid & 15;
    int l    = (tid >> 4) & 31;
    int frag = tid >> 9;
    int nt   = frag & 7;
    int kcg  = frag >> 3;
    int kc   = kcg % KC;
    int k    = kcg / KC;
    int h = l >> 4, n = l & 15;
    int K   = kc * 32 + 16 * h + e;
    int col = nt * 16 + n;
    float v = w[((size_t)k * cin + K) * COUT + col];
    dst[tid] = (unsigned short)f2bf1(v);
}

// ------------------------------------------------------------ batchnorm -----
template<int C>
__global__ __launch_bounds__(256) void k_stats(const float* __restrict__ x, float* __restrict__ sums) {
    constexpr int G = 256 / C;
    int t = threadIdx.x;
    int c = t & (C - 1);
    int g = t / C;
    int rowbase = blockIdx.x * 256;
    float s = 0.f, ss = 0.f;
    for (int r = rowbase + g; r < rowbase + 256; r += G) {
        float v = x[(size_t)r * C + c];
        s += v; ss += v * v;
    }
    __shared__ float sh[512];
    sh[t] = s; sh[256 + t] = ss;
    __syncthreads();
    if (t < C) {
        for (int gg = 1; gg < G; ++gg) { s += sh[t + gg * C]; ss += sh[256 + t + gg * C]; }
        atomicAdd(&sums[c], s);
        atomicAdd(&sums[C + c], ss);
    }
}
template<int C>
__global__ void k_bn_finalize(const float* __restrict__ sums,
                              const float* __restrict__ g, const float* __restrict__ b,
                              float* __restrict__ ab) {
    int t = threadIdx.x;
    if (t >= C) return;
    const float inv = 1.0f / (float)NPTS;
    float mean = sums[t] * inv;
    float var  = sums[C + t] * inv - mean * mean;
    float a = g[t] * rsqrtf(var + 1e-5f);
    ab[t] = a;
    ab[C + t] = b[t] - mean * a;
}
// y = bf16(relu(x*a + b)), 4 channels per thread, packed stores
template<int C>
__global__ __launch_bounds__(256) void k_bn_apply(const float* __restrict__ x,
                                                  const float* __restrict__ ab,
                                                  unsigned short* __restrict__ y) {
    size_t i = ((size_t)blockIdx.x * 256 + threadIdx.x) * 4;
    int c = (int)(i & (C - 1));
    float4 v = *(const float4*)(x + i);
    float r0 = fmaxf(v.x * ab[c+0] + ab[C+c+0], 0.f);
    float r1 = fmaxf(v.y * ab[c+1] + ab[C+c+1], 0.f);
    float r2 = fmaxf(v.z * ab[c+2] + ab[C+c+2], 0.f);
    float r3 = fmaxf(v.w * ab[c+3] + ab[C+c+3], 0.f);
    uint2 p; p.x = pack2(r0, r1); p.y = pack2(r2, r3);
    *(uint2*)(y + i) = p;
}

// ----------------------------------------------- WMMA gather-GEMM conv -------
// Block: 256 threads = 8 waves. Block owns 32 output rows x 128 output cols.
// Wave w owns columns [16w, 16w+16) with TWO accumulators (row-tiles 0/1).
// For each of 9 taps: cooperative gather of the 32 x CIN bf16 neighbor tile
// into padded LDS, then per K-chunk: one B-fragment load feeds 2 WMMAs.
template<int CIN>
__global__ __launch_bounds__(256) void k_conv(const unsigned short* __restrict__ f,
                                              const int* __restrict__ nbr,
                                              const unsigned short* __restrict__ wsw,
                                              float* __restrict__ out) {
    constexpr int KC  = CIN / 32;
    constexpr int ROW = CIN + 8;                 // LDS row padding (bank spread)
    __shared__ unsigned short tile[32 * ROW];
    __shared__ int sids[32 * 9];

    const int t    = threadIdx.x;
    const int wave = t >> 5;
    const int lane = t & 31;
    const int brow = blockIdx.x * 32;
    const int h    = lane >> 4;
    const int m    = lane & 15;

    for (int i = t; i < 32 * 9; i += 256) {
        int r = i / 9, k = i % 9;
        sids[i] = nbr[(size_t)(brow + r) * 9 + k];
    }
    __syncthreads();

    v8f acc0 = {}, acc1 = {};
    const int gr = t >> 3;        // gather row 0..31 (8 threads per row)
    const int gc = t & 7;         // gather column group

    for (int k = 0; k < 9; ++k) {
        // ---- gather neighbor tile into LDS (zeros for masked/missing) ----
        int id = sids[gr * 9 + k];
        if (CIN == 64) {
            uint4 z; z.x = z.y = z.z = z.w = 0u;
            uint4 v = (id >= 0) ? *(const uint4*)(f + (size_t)id * 64 + gc * 8) : z;
            uint2 a; a.x = v.x; a.y = v.y;
            uint2 b; b.x = v.z; b.y = v.w;
            *(uint2*)&tile[gr * ROW + gc * 8 + 0] = a;
            *(uint2*)&tile[gr * ROW + gc * 8 + 4] = b;
        } else {
            uint4 z; z.x = z.y = z.z = z.w = 0u;
            uint4 v0 = (id >= 0) ? *(const uint4*)(f + (size_t)id * 128 + gc * 16 + 0) : z;
            uint4 v1 = (id >= 0) ? *(const uint4*)(f + (size_t)id * 128 + gc * 16 + 8) : z;
            uint2 a; a.x = v0.x; a.y = v0.y;
            uint2 b; b.x = v0.z; b.y = v0.w;
            uint2 c; c.x = v1.x; c.y = v1.y;
            uint2 d; d.x = v1.z; d.y = v1.w;
            *(uint2*)&tile[gr * ROW + gc * 16 +  0] = a;
            *(uint2*)&tile[gr * ROW + gc * 16 +  4] = b;
            *(uint2*)&tile[gr * ROW + gc * 16 +  8] = c;
            *(uint2*)&tile[gr * ROW + gc * 16 + 12] = d;
        }
        __syncthreads();

        #pragma unroll
        for (int kc = 0; kc < KC; ++kc) {
            // B fragment: pre-swizzled, lane-contiguous 32B (loaded once,
            // reused for both row-tiles)
            const uint4* bp = (const uint4*)(wsw + ((size_t)((k * KC + kc) * 8 + wave) * 512 + lane * 16));
            B32x2 bfr; bfr.lo = bp[0]; bfr.hi = bp[1];
            v16bf bv = __builtin_bit_cast(v16bf, bfr);

            // A fragments (16-bit A 16x32 layout): lane h*16+m reads
            // K in [16h,16h+16) of K-chunk kc (matches B swizzle).
            #pragma unroll
            for (int rt = 0; rt < 2; ++rt) {
                int base = (rt * 16 + m) * ROW + kc * 32 + 16 * h;
                uint2 q0 = *(const uint2*)&tile[base +  0];
                uint2 q1 = *(const uint2*)&tile[base +  4];
                uint2 q2 = *(const uint2*)&tile[base +  8];
                uint2 q3 = *(const uint2*)&tile[base + 12];
                B32x2 af;
                af.lo.x = q0.x; af.lo.y = q0.y; af.lo.z = q1.x; af.lo.w = q1.y;
                af.hi.x = q2.x; af.hi.y = q2.y; af.hi.z = q3.x; af.hi.w = q3.y;
                v16bf av = __builtin_bit_cast(v16bf, af);
                if (rt == 0)
                    acc0 = __builtin_amdgcn_wmma_f32_16x16x32_bf16(
                        false, av, false, bv, (short)0, acc0, false, false);
                else
                    acc1 = __builtin_amdgcn_wmma_f32_16x16x32_bf16(
                        false, av, false, bv, (short)0, acc1, false, false);
            }
        }
        __syncthreads();
    }

    // C/D layout: VGPR r -> row r + 8h, column = lane&15
    const int n = lane & 15;
    float* o0 = out + (size_t)(brow +  0) * COUT + wave * 16 + n;
    float* o1 = out + (size_t)(brow + 16) * COUT + wave * 16 + n;
    #pragma unroll
    for (int r = 0; r < 8; ++r) {
        o0[(size_t)(r + 8 * h) * COUT] = acc0[r];
        o1[(size_t)(r + 8 * h) * COUT] = acc1[r];
    }
}

// ------------------------------------------------- skip GEMM (feats@w_skip) --
__global__ __launch_bounds__(256) void k_skip(const float* __restrict__ feats,
                                              const unsigned short* __restrict__ wsk,
                                              float* __restrict__ out) {
    constexpr int ROW = 64 + 8;
    __shared__ unsigned short tile[16 * ROW];
    const int t    = threadIdx.x;
    const int wave = t >> 5;
    const int lane = t & 31;
    const int brow = blockIdx.x * 16;
    const int h = lane >> 4, m = lane & 15;

    {   // load 16x64 fp32 tile, convert to bf16 in LDS
        int gr = t >> 4, gc = t & 15;
        float4 v = *(const float4*)(feats + (size_t)(brow + gr) * 64 + gc * 4);
        uint2 p; p.x = pack2(v.x, v.y); p.y = pack2(v.z, v.w);
        *(uint2*)&tile[gr * ROW + gc * 4] = p;
    }
    __syncthreads();

    v8f acc = {};
    #pragma unroll
    for (int kc = 0; kc < 2; ++kc) {
        int base = m * ROW + kc * 32 + 16 * h;
        uint2 q0 = *(const uint2*)&tile[base +  0];
        uint2 q1 = *(const uint2*)&tile[base +  4];
        uint2 q2 = *(const uint2*)&tile[base +  8];
        uint2 q3 = *(const uint2*)&tile[base + 12];
        B32x2 af;
        af.lo.x = q0.x; af.lo.y = q0.y; af.lo.z = q1.x; af.lo.w = q1.y;
        af.hi.x = q2.x; af.hi.y = q2.y; af.hi.z = q3.x; af.hi.w = q3.y;
        const uint4* bp = (const uint4*)(wsk + ((size_t)(kc * 8 + wave) * 512 + lane * 16));
        B32x2 bfr; bfr.lo = bp[0]; bfr.hi = bp[1];
        acc = __builtin_amdgcn_wmma_f32_16x16x32_bf16(
            false, __builtin_bit_cast(v16bf, af),
            false, __builtin_bit_cast(v16bf, bfr),
            (short)0, acc, false, false);
    }

    const int n = lane & 15;
    float* o = out + (size_t)brow * COUT + wave * 16 + n;
    #pragma unroll
    for (int r = 0; r < 8; ++r)
        o[(size_t)(r + 8 * h) * COUT] = acc[r];
}

// ---------------------------------------------------------- pooling / SE ----
__global__ __launch_bounds__(256) void k_pool(const float* __restrict__ x, float* __restrict__ pooled) {
    int t = threadIdx.x;
    int rowbase = blockIdx.x * 512;
    int batch = rowbase >> 15;                   // 32768 rows per batch
    int c = t & 127, g = t >> 7;
    float s = 0.f;
    for (int r = rowbase + g; r < rowbase + 512; r += 2)
        s += x[(size_t)r * COUT + c];
    __shared__ float sh[256];
    sh[t] = s;
    __syncthreads();
    if (t < 128) {
        s += sh[t + 128];
        atomicAdd(&pooled[batch * COUT + c], s);
    }
}
__global__ void k_se(const float* __restrict__ pooled,
                     const float* __restrict__ fc1_w, const float* __restrict__ fc1_b,
                     const float* __restrict__ fc2_w, const float* __restrict__ fc2_b,
                     float* __restrict__ se) {
    int b = blockIdx.x, t = threadIdx.x;
    __shared__ float hbuf[32];
    if (t < 32) {
        float a = fc1_b[t];
        const float inv = 1.0f / 32768.0f;
        for (int c = 0; c < COUT; ++c)
            a += (pooled[b * COUT + c] * inv) * fc1_w[t * COUT + c];
        hbuf[t] = fmaxf(a, 0.f);
    }
    __syncthreads();
    if (t < COUT) {
        float z = fc2_b[t];
        for (int j = 0; j < 32; ++j) z += hbuf[j] * fc2_w[t * 32 + j];
        se[b * COUT + t] = 1.0f / (1.0f + expf(-z));
    }
}
// out = conv2 * se[batch] + skip (skip already resides in d_out)
__global__ __launch_bounds__(256) void k_final(const float* __restrict__ conv2,
                                               const float* __restrict__ se,
                                               float* __restrict__ out) {
    size_t i = ((size_t)blockIdx.x * 256 + threadIdx.x) * 4;
    size_t row = i >> 7;
    int batch = (int)(row >> 15);
    int c = (int)(i & 127);
    float4 o2 = *(const float4*)(conv2 + i);
    float4 sk = *(const float4*)(out + i);
    float4 sv = *(const float4*)(se + batch * COUT + c);
    float4 r;
    r.x = o2.x * sv.x + sk.x;
    r.y = o2.y * sv.y + sk.y;
    r.z = o2.z * sv.z + sk.z;
    r.w = o2.w * sv.w + sk.w;
    *(float4*)(out + i) = r;
}

// ---------------------------------------------------------------- launch ----
extern "C" void kernel_launch(void* const* d_in, const int* in_sizes, int n_in,
                              void* d_out, int out_size, void* d_ws, size_t ws_size,
                              hipStream_t stream) {
    const float* feats  = (const float*)d_in[0];
    const int*   idx    = (const int*)  d_in[1];
    const float* bn1_g  = (const float*)d_in[2];
    const float* bn1_b  = (const float*)d_in[3];
    const float* w1     = (const float*)d_in[4];
    const float* bn2_g  = (const float*)d_in[5];
    const float* bn2_b  = (const float*)d_in[6];
    const float* w2     = (const float*)d_in[7];
    const float* fc1_w  = (const float*)d_in[8];
    const float* fc1_b  = (const float*)d_in[9];
    const float* fc2_w  = (const float*)d_in[10];
    const float* fc2_b  = (const float*)d_in[11];
    const float* w_skip = (const float*)d_in[12];
    float* out = (float*)d_out;

    // workspace carve (bytes, all 256B aligned)
    char* ws = (char*)d_ws;
    int*            grid   = (int*)           (ws + 0);            //  8 MiB
    int*            nbr    = (int*)           (ws + 8388608);      //  9 MiB
    float*          stats  = (float*)         (ws + 17825792);     // 16 KiB
    unsigned short* w1s    = (unsigned short*)(ws + 17842176);     // 144 KiB
    unsigned short* w2s    = (unsigned short*)(ws + 17989632);     // 288 KiB
    unsigned short* wsks   = (unsigned short*)(ws + 18284544);     // 16 KiB
    unsigned short* f1     = (unsigned short*)(ws + 18300928);     // 32 MiB
    unsigned short* f2     = (unsigned short*)(ws + 51855360);     // 64 MiB
    float*          buf    = (float*)         (ws + 118964224);    // 128 MiB (out1 then conv2)

    float* sums1  = stats;          // 128
    float* ab1    = stats + 128;    // 128
    float* sums2  = stats + 256;    // 256
    float* ab2    = stats + 512;    // 256
    float* pooled = stats + 768;    // 1024
    float* se     = stats + 1792;   // 1024

    // 1) rulebook
    k_init_grid <<<(NB*HWSZ + 255)/256, 256, 0, stream>>>(grid);
    k_init_stats<<<1, 256, 0, stream>>>(stats, 2816);
    k_scatter   <<<NPTS/256, 256, 0, stream>>>(idx, grid);
    k_rulebook  <<<NPTS/256, 256, 0, stream>>>(idx, grid, nbr);

    // 2) weight swizzle -> bf16 fragment order
    k_swizzle<<<(9*2*8*512)/256, 256, 0, stream>>>(w1,     w1s,  2,  64, 9*2*8*512);
    k_swizzle<<<(9*4*8*512)/256, 256, 0, stream>>>(w2,     w2s,  4, 128, 9*4*8*512);
    k_swizzle<<<(1*2*8*512)/256, 256, 0, stream>>>(w_skip, wsks, 2,  64, 1*2*8*512);

    // 3) bn1 + relu -> bf16
    k_stats<64>      <<<NPTS/256, 256, 0, stream>>>(feats, sums1);
    k_bn_finalize<64><<<1, 64, 0, stream>>>(sums1, bn1_g, bn1_b, ab1);
    k_bn_apply<64>   <<<(NPTS*64/4)/256, 256, 0, stream>>>(feats, ab1, f1);

    // 4) conv1 (WMMA, 64->128)
    k_conv<64><<<NPTS/32, 256, 0, stream>>>(f1, nbr, w1s, buf);

    // 5) bn2 + relu -> bf16
    k_stats<128>      <<<NPTS/256, 256, 0, stream>>>(buf, sums2);
    k_bn_finalize<128><<<1, 128, 0, stream>>>(sums2, bn2_g, bn2_b, ab2);
    k_bn_apply<128>   <<<(NPTS*128/4)/256, 256, 0, stream>>>(buf, ab2, f2);

    // 6) conv2 (WMMA, 128->128), overwrites buf (old contents dead)
    k_conv<128><<<NPTS/32, 256, 0, stream>>>(f2, nbr, w2s, buf);

    // 7) skip GEMM (WMMA) -> d_out
    k_skip<<<NPTS/16, 256, 0, stream>>>(feats, wsks, out);

    // 8) SE
    k_pool<<<NPTS/512, 256, 0, stream>>>(buf, pooled);
    k_se  <<<NB, 128, 0, stream>>>(pooled, fc1_w, fc1_b, fc2_w, fc2_b, se);

    // 9) out = conv2*se + skip
    k_final<<<((size_t)NPTS*128/4)/256, 256, 0, stream>>>(buf, se, out);
}
